// Net_74259984548321
// MI455X (gfx1250) — matrix operand order repaired and checked by reading, not compiled, not run on hardware
//
#include <hip/hip_runtime.h>
#include <hip/hip_bf16.h>
#include <math.h>

// Problem constants (from the reference)
#define BB    64      // batch
#define TENC  80      // encoder frames
#define FEAT  4096
#define HH    256     // hidden
#define VV    8000    // vocab
#define LL    32      // decoder length
#define G4H   (4*HH)  // 1024 gate width

typedef __attribute__((ext_vector_type(16))) _Float16 v16h;
typedef __attribute__((ext_vector_type(8)))  _Float16 v8h;
typedef __attribute__((ext_vector_type(8)))  float    v8f;
typedef __attribute__((ext_vector_type(4)))  float    v4f;

// ---------------------------------------------------------------------------
// Utility kernels
// ---------------------------------------------------------------------------
__global__ void f32_to_f16_kernel(const float* __restrict__ src,
                                  _Float16* __restrict__ dst, int n) {
  int i = blockIdx.x * blockDim.x + threadIdx.x;
  if (i < n) dst[i] = (_Float16)src[i];
}

__global__ void zero_f32_kernel(float* __restrict__ p, int n) {
  int i = blockIdx.x * blockDim.x + threadIdx.x;
  if (i < n) p[i] = 0.0f;
}

// argmax over last dim (first-max tie break, like jnp.argmax)
__global__ __launch_bounds__(256) void argmax_rows_kernel(
    const float* __restrict__ x, int cols, int* __restrict__ out) {
  __shared__ float sv[256];
  __shared__ int   si[256];
  int r = blockIdx.x, tid = threadIdx.x;
  const float* row = x + (size_t)r * cols;
  float best = -3.4e38f; int bi = 0;
  for (int c = tid; c < cols; c += 256) {
    float v = row[c];
    if (v > best) { best = v; bi = c; }
  }
  sv[tid] = best; si[tid] = bi;
  __syncthreads();
  for (int s = 128; s > 0; s >>= 1) {
    if (tid < s) {
      if (sv[tid + s] > sv[tid] ||
          (sv[tid + s] == sv[tid] && si[tid + s] < si[tid])) {
        sv[tid] = sv[tid + s]; si[tid] = si[tid + s];
      }
    }
    __syncthreads();
  }
  if (tid == 0) out[r] = si[0];
}

// ---------------------------------------------------------------------------
// Generic fused-gates GEMM with CDNA5 WMMA, 1x4 register blocking:
//   C[M,N](f32) = A1[M,K1](f32->f16) @ W1[N,K1]^T (f16)
//              (+ A2[M,K2] @ W2[N,K2]^T) (+ bias[N]) (+ pre[m*preStride + n])
// One wave computes one 16x64 strip of C (4 accumulators); one 16x32 A
// fragment is converted once and feeds 4 v_wmma_f32_16x16x32_f16 per K-block.
// Fragment layouts follow CDNA5 ISA:
//   A: lane(M) half-waves hold K {0-7,16-23}/{8-15,24-31}
//   B: lane(N) half-waves hold K {0-7,16-23}/{8-15,24-31}
//   C/D: 8 f32 VGPRs, M = v + 8*(lane>=16), N = lane&15
// ---------------------------------------------------------------------------
__device__ __forceinline__ void wmma_k_loop4(v8f acc[4],
    const float* __restrict__ A, int lda,
    const _Float16* __restrict__ W, int ldw, int K,
    int arow, int bcol0, int khalf) {
  const float* ap = A + (size_t)arow * lda + khalf * 8;
  const _Float16* wp[4];
#pragma unroll
  for (int j = 0; j < 4; ++j)
    wp[j] = W + (size_t)(bcol0 + 16 * j) * ldw + khalf * 8;

  for (int kb = 0; kb < K; kb += 32) {
    // A fragment: load f32, convert to f16 in-register (once per 4 WMMAs)
    v4f a0 = *(const v4f*)(ap + kb);
    v4f a1 = *(const v4f*)(ap + kb + 4);
    v4f a2 = *(const v4f*)(ap + kb + 16);
    v4f a3 = *(const v4f*)(ap + kb + 20);
    v16h av;
#pragma unroll
    for (int i = 0; i < 4; ++i) {
      av[i]      = (_Float16)a0[i];
      av[i + 4]  = (_Float16)a1[i];
      av[i + 8]  = (_Float16)a2[i];
      av[i + 12] = (_Float16)a3[i];
    }
#pragma unroll
    for (int j = 0; j < 4; ++j) {
      v8h b0 = *(const v8h*)(wp[j] + kb);
      v8h b1 = *(const v8h*)(wp[j] + kb + 16);
      v16h bv;
#pragma unroll
      for (int i = 0; i < 8; ++i) { bv[i] = b0[i]; bv[i + 8] = b1[i]; }
      acc[j] = __builtin_amdgcn_wmma_f32_16x16x32_f16(
          /*neg_a=*/false, av, /*neg_b=*/false, bv,
          /*c_mod=*/(short)0, acc[j], /*reuse_a=*/false, /*reuse_b=*/false);
    }
  }
}

__global__ __launch_bounds__(256) void gates_gemm_kernel(
    const float* __restrict__ A1, int lda1,
    const _Float16* __restrict__ W1, int ldw1, int K1,
    const float* __restrict__ A2, int lda2,
    const _Float16* __restrict__ W2, int ldw2, int K2,
    const float* __restrict__ bias,
    const float* __restrict__ pre, int preStride,
    float* __restrict__ C, int M, int N) {
  const int lane  = threadIdx.x & 31;
  const int wave  = threadIdx.x >> 5;
  const int nSN   = N >> 6;                  // 16x64 supertiles along N
  const int nTile = (M >> 4) * nSN;
  int tile = blockIdx.x * 8 + wave;          // 8 waves / block
  if (tile >= nTile) return;                 // wave-uniform: EXEC stays all-1s
  const int tm    = (tile / nSN) << 4;
  const int tn0   = (tile % nSN) << 6;
  const int l15   = lane & 15;
  const int khalf = lane >> 4;
  const int arow  = tm + l15;                // A row owned by this lane
  const int bcol0 = tn0 + l15;               // first output column of strip

  v8f acc[4] = {};
  wmma_k_loop4(acc, A1, lda1, W1, ldw1, K1, arow, bcol0, khalf);
  if (K2 > 0)
    wmma_k_loop4(acc, A2, lda2, W2, ldw2, K2, arow, bcol0, khalf);

#pragma unroll
  for (int j = 0; j < 4; ++j) {
    int cidx = tn0 + 16 * j + l15;
    float badd = bias ? bias[cidx] : 0.0f;
#pragma unroll
    for (int v = 0; v < 8; ++v) {
      int r = tm + v + khalf * 8;
      float val = acc[j][v] + badd;
      if (pre) val += pre[(size_t)r * preStride + cidx];
      C[(size_t)r * N + cidx] = val;
    }
  }
}

// ---------------------------------------------------------------------------
// LSTM gate nonlinearity + state update.  gates = [i | f | g | o] (each H).
// ---------------------------------------------------------------------------
__global__ __launch_bounds__(256) void lstm_update_kernel(
    const float* __restrict__ gates,
    float* __restrict__ h, float* __restrict__ c,
    float* __restrict__ hcopy, int hcopyStride) {
  int idx = blockIdx.x * blockDim.x + threadIdx.x;
  if (idx >= BB * HH) return;
  int m = idx >> 8;            // / HH
  int j = idx & (HH - 1);
  const float* g = gates + (size_t)m * G4H;
  float gi = g[j], gf = g[j + HH], gg = g[j + 2 * HH], go = g[j + 3 * HH];
  float si = 1.0f / (1.0f + expf(-gi));
  float sf = 1.0f / (1.0f + expf(-gf));
  float so = 1.0f / (1.0f + expf(-go));
  float cn = sf * c[idx] + si * tanhf(gg);
  c[idx] = cn;
  float hn = so * tanhf(cn);
  h[idx] = hn;
  if (hcopy) hcopy[(size_t)m * hcopyStride + j] = hn;
}

// ---------------------------------------------------------------------------
// Cross-entropy for one decoder step: loss += mean_b(lse - logit[tgt]) / B
// One block per batch row; atomicAdd the scalar contribution.
// ---------------------------------------------------------------------------
__global__ __launch_bounds__(256) void ce_kernel(
    const float* __restrict__ logits, const int* __restrict__ tgt,
    int tIdx, float* __restrict__ out) {
  __shared__ float red[256];
  int b = blockIdx.x, tid = threadIdx.x;
  const float* row = logits + (size_t)b * VV;
  float mx = -3.4e38f;
  for (int c = tid; c < VV; c += 256) mx = fmaxf(mx, row[c]);
  red[tid] = mx;
  __syncthreads();
  for (int s = 128; s > 0; s >>= 1) {
    if (tid < s) red[tid] = fmaxf(red[tid], red[tid + s]);
    __syncthreads();
  }
  mx = red[0];
  __syncthreads();
  float sum = 0.0f;
  for (int c = tid; c < VV; c += 256) sum += expf(row[c] - mx);
  red[tid] = sum;
  __syncthreads();
  for (int s = 128; s > 0; s >>= 1) {
    if (tid < s) red[tid] += red[tid + s];
    __syncthreads();
  }
  if (tid == 0) {
    float lse = logf(red[0]) + mx;
    int t = tgt[b * LL + tIdx];
    atomicAdd(out, (lse - row[t]) / (float)(BB * BB));
  }
}

// ---------------------------------------------------------------------------
// Attention: attn = softmax_t(h2_list[b,t,:].h2[b,:]); h2[b,:] = attn @ h2_list
// One block (256 threads == H) per batch element.
// ---------------------------------------------------------------------------
__global__ __launch_bounds__(256) void attention_kernel(
    const float* __restrict__ h2_list, float* __restrict__ h2) {
  __shared__ float hv[HH];
  __shared__ float sc[TENC];
  int b = blockIdx.x, tid = threadIdx.x;
  hv[tid] = h2[b * HH + tid];
  __syncthreads();
  if (tid < TENC) {
    const float* row = h2_list + (size_t)b * TENC * HH + (size_t)tid * HH;
    float d = 0.0f;
    for (int j = 0; j < HH; ++j) d += row[j] * hv[j];
    sc[tid] = d;
  }
  __syncthreads();
  if (tid == 0) {
    float mx = sc[0];
    for (int t = 1; t < TENC; ++t) mx = fmaxf(mx, sc[t]);
    float s = 0.0f;
    for (int t = 0; t < TENC; ++t) { float e = expf(sc[t] - mx); sc[t] = e; s += e; }
    float inv = 1.0f / s;
    for (int t = 0; t < TENC; ++t) sc[t] *= inv;
  }
  __syncthreads();
  float acc = 0.0f;
  const float* base = h2_list + (size_t)b * TENC * HH + tid;
  for (int t = 0; t < TENC; ++t) acc += sc[t] * base[(size_t)t * HH];
  h2[b * HH + tid] = acc;
}

// ---------------------------------------------------------------------------
// Host-side orchestration (graph-capturable: only launches on `stream`)
// ---------------------------------------------------------------------------
extern "C" void kernel_launch(void* const* d_in, const int* in_sizes, int n_in,
                              void* d_out, int out_size, void* d_ws, size_t ws_size,
                              hipStream_t stream) {
  (void)in_sizes; (void)n_in; (void)out_size; (void)ws_size;

  const float* feat    = (const float*)d_in[0];
  const float* caption = (const float*)d_in[1];
  const float* cap1h   = (const float*)d_in[2];
  // d_in[3] keep_rate == 1 (dropout dead)
  const float* e1_Wih  = (const float*)d_in[4];
  const float* e1_Whh  = (const float*)d_in[5];
  const float* e1_b    = (const float*)d_in[6];
  const float* e2_Wih  = (const float*)d_in[7];
  const float* e2_Whh  = (const float*)d_in[8];
  const float* e2_b    = (const float*)d_in[9];
  // d_in[10] d1_Wih unused (decoder layer-1 input is zeros)
  const float* d1_Whh  = (const float*)d_in[11];
  const float* d1_b    = (const float*)d_in[12];
  const float* d2_Wih  = (const float*)d_in[13];
  const float* d2_Whh  = (const float*)d_in[14];
  const float* d2_b    = (const float*)d_in[15];
  const float* out_W   = (const float*)d_in[16];
  const float* out_b   = (const float*)d_in[17];

  // ---- workspace bump allocator (256B aligned) ----
  char* ws = (char*)d_ws;
  size_t off = 0;
  auto alloc = [&](size_t bytes) -> void* {
    off = (off + 255) & ~(size_t)255;
    void* p = ws + off;
    off += bytes;
    return p;
  };

  _Float16* w_e1_Wih = (_Float16*)alloc((size_t)G4H * FEAT  * 2);
  _Float16* w_e1_Whh = (_Float16*)alloc((size_t)G4H * HH    * 2);
  _Float16* w_e2_Wih = (_Float16*)alloc((size_t)G4H * 2*HH  * 2);
  _Float16* w_e2_Whh = (_Float16*)alloc((size_t)G4H * HH    * 2);
  _Float16* w_d1_Whh = (_Float16*)alloc((size_t)G4H * HH    * 2);
  _Float16* w_d2_Wih = (_Float16*)alloc((size_t)G4H * 2*HH  * 2);
  _Float16* w_d2_Whh = (_Float16*)alloc((size_t)G4H * HH    * 2);
  _Float16* w_outW   = (_Float16*)alloc((size_t)VV  * HH    * 2);

  float* enc1pre = (float*)alloc((size_t)BB * TENC * G4H * 4); // feat@Wih.T + b
  float* capproj = (float*)alloc((size_t)BB * LL   * G4H * 4); // cap@Wih_l.T + b
  float* gates   = (float*)alloc((size_t)BB * G4H * 4);
  float* logits  = (float*)alloc((size_t)BB * VV  * 4);
  float* states  = (float*)alloc((size_t)4 * BB * HH * 4);
  float* h2_list = (float*)alloc((size_t)BB * TENC * HH * 4);
  int*   tgt     = (int*)alloc((size_t)BB * LL * 4);

  float* h1 = states;
  float* c1 = states + 1 * BB * HH;
  float* h2 = states + 2 * BB * HH;
  float* c2 = states + 3 * BB * HH;

  auto conv = [&](const float* s, _Float16* d, int n) {
    f32_to_f16_kernel<<<(n + 255) / 256, 256, 0, stream>>>(s, d, n);
  };
  conv(e1_Wih, w_e1_Wih, G4H * FEAT);
  conv(e1_Whh, w_e1_Whh, G4H * HH);
  conv(e2_Wih, w_e2_Wih, G4H * 2 * HH);
  conv(e2_Whh, w_e2_Whh, G4H * HH);
  conv(d1_Whh, w_d1_Whh, G4H * HH);
  conv(d2_Wih, w_d2_Wih, G4H * 2 * HH);
  conv(d2_Whh, w_d2_Whh, G4H * HH);
  conv(out_W,  w_outW,   VV * HH);

  zero_f32_kernel<<<(4 * BB * HH + 255) / 256, 256, 0, stream>>>(states, 4 * BB * HH);
  zero_f32_kernel<<<1, 32, 0, stream>>>((float*)d_out, 1);

  argmax_rows_kernel<<<BB * LL, 256, 0, stream>>>(cap1h, VV, tgt);

  auto gemm = [&](const float* A1, int lda1, const _Float16* W1, int ldw1, int K1,
                  const float* A2, int lda2, const _Float16* W2, int ldw2, int K2,
                  const float* bias, const float* pre, int preStride,
                  float* C, int M, int N) {
    int tiles = (M / 16) * (N / 64);   // 16x64 supertiles, one per wave
    gates_gemm_kernel<<<(tiles + 7) / 8, 256, 0, stream>>>(
        A1, lda1, W1, ldw1, K1, A2, lda2, W2, ldw2, K2,
        bias, pre, preStride, C, M, N);
  };

  // ---- hoisted input projections (the bulk of the FLOPs) ----
  // enc1pre[b*T+t, :] = feat[b,t,:] @ e1_Wih.T + e1_b
  gemm(feat, FEAT, w_e1_Wih, FEAT, FEAT, nullptr, 0, nullptr, 0, 0,
       e1_b, nullptr, 0, enc1pre, BB * TENC, G4H);
  // capproj[b*L+t, :] = caption[b,t,:] @ d2_Wih[:, :H].T + d2_b
  gemm(caption, HH, w_d2_Wih, 2 * HH, HH, nullptr, 0, nullptr, 0, 0,
       d2_b, nullptr, 0, capproj, BB * LL, G4H);

  // ---- encoder: 80 recurrent steps ----
  for (int t = 0; t < TENC; ++t) {
    // e1 gates = h1 @ Whh.T + (precomputed x-proj + bias)
    gemm(h1, HH, w_e1_Whh, HH, HH, nullptr, 0, nullptr, 0, 0,
         nullptr, enc1pre + (size_t)t * G4H, TENC * G4H, gates, BB, G4H);
    lstm_update_kernel<<<(BB * HH + 255) / 256, 256, 0, stream>>>(gates, h1, c1, nullptr, 0);
    // e2 input = [zeros(H), h1] -> only right half of e2_Wih contributes
    gemm(h1, HH, w_e2_Wih + HH, 2 * HH, HH, h2, HH, w_e2_Whh, HH, HH,
         e2_b, nullptr, 0, gates, BB, G4H);
    lstm_update_kernel<<<(BB * HH + 255) / 256, 256, 0, stream>>>(
        gates, h2, c2, h2_list + (size_t)t * HH, TENC * HH);
  }

  // ---- decoder: 31 steps ----
  for (int t = 0; t < LL - 1; ++t) {
    // d1 input is zero -> gates = h1 @ d1_Whh.T + d1_b
    gemm(h1, HH, w_d1_Whh, HH, HH, nullptr, 0, nullptr, 0, 0,
         d1_b, nullptr, 0, gates, BB, G4H);
    lstm_update_kernel<<<(BB * HH + 255) / 256, 256, 0, stream>>>(gates, h1, c1, nullptr, 0);
    // d2 input = [cap_t, h1] -> precomputed cap-proj (incl. bias) + right half Wih
    gemm(h1, HH, w_d2_Wih + HH, 2 * HH, HH, h2, HH, w_d2_Whh, HH, HH,
         nullptr, capproj + (size_t)t * G4H, LL * G4H, gates, BB, G4H);
    lstm_update_kernel<<<(BB * HH + 255) / 256, 256, 0, stream>>>(gates, h2, c2, nullptr, 0);
    // vocab logits
    gemm(h2, HH, w_outW, HH, HH, nullptr, 0, nullptr, 0, 0,
         out_b, nullptr, 0, logits, BB, VV);
    ce_kernel<<<BB, 256, 0, stream>>>(logits, tgt, t + 1, (float*)d_out);
    // attention replaces h2 with attended context
    attention_kernel<<<BB, 256, 0, stream>>>(h2_list, h2);
  }
}
